// MoESwiGLU_36481452213063
// MI455X (gfx1250) — matrix operand order repaired and checked by reading, not compiled, not run on hardware
//
#include <hip/hip_runtime.h>

// ---------------------------------------------------------------------------
// MoE SwiGLU for MI455X (gfx1250): wave32, WMMA bf16 (v_wmma_f32_16x16x32_bf16)
// ---------------------------------------------------------------------------

typedef __attribute__((ext_vector_type(16))) __bf16 v16bf;
typedef __attribute__((ext_vector_type(8)))  float  v8f;

constexpr int kH = 1024;
constexpr int kE = 8;
constexpr int kI = 2816;
constexpr int kT = 16384;           // 4 * 4096 tokens

constexpr int MT = 32;              // tokens per block tile
constexpr int IC = 128;             // I-chunk per fused iteration
constexpr int KC = 64;              // H sub-chunk staged per stage-1 step
constexpr int TILES_PER_E = kT / MT;   // 512

// ---- LDS layout (bytes); stage-1 scratch and stage-2 w2 staging are unioned
constexpr int TOK_OFF   = 0;                          // int[32]
constexpr int TWT_OFF   = 128;                        // float[32]
constexpr int HS_OFF    = 256;                        // 32 rows x 272B (128 bf16 + pad)
constexpr int HS_STRIDE = 272;                        // 17*16 -> 16B aligned rows
constexpr int SCR_OFF   = HS_OFF + MT * HS_STRIDE;    // 8960
constexpr int XS_OFF    = SCR_OFF;                    // 32 rows x 144B (64 bf16 + pad)
constexpr int XS_STRIDE = 144;
constexpr int W1_OFF    = XS_OFF + MT * XS_STRIDE;    // 13568 : 128 c-rows x 144B
constexpr int WT_STRIDE = 144;
constexpr int W3_OFF    = W1_OFF + IC * WT_STRIDE;    // 32000
constexpr int W2_OFF    = SCR_OFF;                    // 512 c-rows x 80B (32 bf16 + pad)
constexpr int W2_STRIDE = 80;
constexpr int SMEM_BYTES = W3_OFF + IC * WT_STRIDE;   // 50432 (< 64KB static LDS)

// ---- fp32 -> packed bf16x2 conversion (hardware pack, 1 VALU per 2 elems) --
__device__ __forceinline__ unsigned cvt2bf(float lo, float hi) {
#if __has_builtin(__builtin_amdgcn_cvt_pk_bf16_f32)
    typedef __attribute__((ext_vector_type(2))) __bf16 v2bf;
    union { v2bf v; unsigned u; } x;
    x.v = __builtin_amdgcn_cvt_pk_bf16_f32(lo, hi);
    return x.u;
#else
    // v_perm_b32: pack high halves of the two fp32 (truncate-to-bf16)
    union { float f; unsigned u; } a, b;
    a.f = hi; b.f = lo;
    return __builtin_amdgcn_perm(a.u, b.u, 0x07060302u);
#endif
}
__device__ __forceinline__ unsigned short bf1(float f) {
    return (unsigned short)(cvt2bf(f, f) & 0xFFFFu);
}
__device__ __forceinline__ uint4 pack8(const float f[8]) {
    return make_uint4(cvt2bf(f[0], f[1]), cvt2bf(f[2], f[3]),
                      cvt2bf(f[4], f[5]), cvt2bf(f[6], f[7]));
}

struct Frag { union { v16bf v; uint4 u[2]; }; };

// A fragment (16 x 32 bf16, MxK) from row-major LDS [m][k] (pairs contiguous):
//   lanes 0-15: M=lane,    VGPR 0-3 -> K=0..7,  VGPR 4-7 -> K=16..23
//   lanes16-31: M=lane-16, VGPR 0-3 -> K=8..15, VGPR 4-7 -> K=24..31
__device__ __forceinline__ v16bf load_a(const unsigned char* base, int strideB,
                                        int m0, int k0, int lane) {
    Frag f;
    const int m    = m0 + (lane & 15);
    const int koff = (lane >> 4) * 8;
    const unsigned char* p = base + m * strideB + 2 * (k0 + koff);
    f.u[0] = *(const uint4*)(p);        // K = k0+koff   .. +7
    f.u[1] = *(const uint4*)(p + 32);   // K = k0+16+koff.. +7
    return f.v;
}

// B fragment (32 x 16 bf16, KxN) from K-transposed LDS [c][k]:
//   lanes 0-15: N=lane, K=0..7,16..23 ; lanes 16-31: N=lane-16, K=8..15,24..31
__device__ __forceinline__ v16bf load_bt(const unsigned char* base, int strideB,
                                         int c0, int k0, int lane) {
    Frag f;
    const int c    = c0 + (lane & 15);
    const int koff = (lane >> 4) * 8;
    const unsigned char* p = base + c * strideB + 2 * (k0 + koff);
    f.u[0] = *(const uint4*)(p);        // K = k0+koff .. +7
    f.u[1] = *(const uint4*)(p + 32);   // K = k0+16+koff .. +7
    return f.v;
}

__device__ __forceinline__ v8f wmma_bf16(v16bf a, v16bf b, v8f c) {
    return __builtin_amdgcn_wmma_f32_16x16x32_bf16(
        /*neg_a=*/false, a, /*neg_b=*/false, b,
        /*c_mod=*/(short)0, c, /*reuse_a=*/false, /*reuse_b=*/false);
}

// ---------------------------------------------------------------------------
// Kernel 0: zero output + expert counters (harness does not re-poison buffers)
// ---------------------------------------------------------------------------
__global__ void moe_zero_kernel(float* __restrict__ out, int* __restrict__ cnt) {
    const size_t n4 = (size_t)kT * kH / 4;
    const size_t step = (size_t)gridDim.x * blockDim.x;
    float4 z = make_float4(0.f, 0.f, 0.f, 0.f);
    for (size_t i = (size_t)blockIdx.x * blockDim.x + threadIdx.x; i < n4; i += step)
        ((float4*)out)[i] = z;
    if (blockIdx.x == 0 && threadIdx.x < kE) cnt[threadIdx.x] = 0;
}

// ---------------------------------------------------------------------------
// Kernel 1: router — softmax over 8 logits, top-2, renormalize, build lists.
// One wave32 per token.
// ---------------------------------------------------------------------------
__global__ __launch_bounds__(256)
void moe_router_kernel(const float* __restrict__ x, const float* __restrict__ wr,
                       int* __restrict__ cnt, int* __restrict__ lists,
                       float* __restrict__ lw) {
    const int wave = threadIdx.x >> 5;
    const int lane = threadIdx.x & 31;
    const int t = blockIdx.x * 8 + wave;

    const float* xr = x + (size_t)t * kH;
    float acc[kE];
#pragma unroll
    for (int e = 0; e < kE; ++e) acc[e] = 0.f;

    for (int h = lane; h < kH; h += 32) {
        const float xv = xr[h];
        const float* w = wr + h * kE;
        const float4 a = *(const float4*)w;
        const float4 b = *(const float4*)(w + 4);
        acc[0] += xv * a.x; acc[1] += xv * a.y;
        acc[2] += xv * a.z; acc[3] += xv * a.w;
        acc[4] += xv * b.x; acc[5] += xv * b.y;
        acc[6] += xv * b.z; acc[7] += xv * b.w;
    }
#pragma unroll
    for (int off = 16; off > 0; off >>= 1)
#pragma unroll
        for (int e = 0; e < kE; ++e) acc[e] += __shfl_xor(acc[e], off, 32);

    if (lane == 0) {
        float m = acc[0];
#pragma unroll
        for (int e = 1; e < kE; ++e) m = fmaxf(m, acc[e]);
        float p[kE];
#pragma unroll
        for (int e = 0; e < kE; ++e) p[e] = __expf(acc[e] - m);
        int i1 = 0;
#pragma unroll
        for (int e = 1; e < kE; ++e) if (p[e] > p[i1]) i1 = e;
        int i2 = (i1 == 0) ? 1 : 0;
#pragma unroll
        for (int e = 0; e < kE; ++e) if (e != i1 && p[e] > p[i2]) i2 = e;
        const float s = p[i1] + p[i2];
        const float w1p = p[i1] / s, w2p = p[i2] / s;
        int pos = atomicAdd(&cnt[i1], 1);
        lists[i1 * kT + pos] = t; lw[i1 * kT + pos] = w1p;
        pos = atomicAdd(&cnt[i2], 1);
        lists[i2 * kT + pos] = t; lw[i2 * kT + pos] = w2p;
    }
}

// ---------------------------------------------------------------------------
// Kernel 2: fused expert GEMM. Block = 8 waves = 32-token tile of one expert.
// stage1: a1=x@w1, a3=x@w3 (K=1024), h=silu(a1)*a3 -> LDS (bf16)
// stage2: out += h @ w2 (K=128 chunk), epilogue: atomicAdd(weight * acc)
// ---------------------------------------------------------------------------
__global__ __launch_bounds__(256)
void moe_expert_kernel(const float* __restrict__ x,
                       const float* __restrict__ w1g,
                       const float* __restrict__ w3g,
                       const float* __restrict__ w2g,
                       const int* __restrict__ cnt,
                       const int* __restrict__ lists,
                       const float* __restrict__ lw,
                       float* __restrict__ out) {
    const int e    = blockIdx.x >> 9;          // expert (512 tiles each)
    const int tile = blockIdx.x & (TILES_PER_E - 1);
    const int ntok = cnt[e];
    if (tile * MT >= ntok) return;             // uniform early-exit

    __shared__ __align__(16) unsigned char SM[SMEM_BYTES];
    int*   tokS = (int*)(SM + TOK_OFF);
    float* twtS = (float*)(SM + TWT_OFF);

    const int tid  = threadIdx.x;
    const int wave = tid >> 5;
    const int lane = tid & 31;
    const int lh   = lane >> 4;                // lane half
    const int ln   = lane & 15;

    if (tid < MT) {
        const int slot = tile * MT + tid;
        if (slot < ntok) {
            tokS[tid] = lists[e * kT + slot];
            twtS[tid] = lw[e * kT + slot];
        } else {
            tokS[tid] = 0;
            twtS[tid] = 0.f;                   // padded row: never stored
        }
    }
    __syncthreads();

    const float* w1e = w1g + (size_t)e * kH * kI;
    const float* w3e = w3g + (size_t)e * kH * kI;
    const float* w2e = w2g + (size_t)e * kI * kH;

    v8f accO[8][2];                            // [half*4+nt][mt] : wave's 128 H-cols
#pragma unroll
    for (int a = 0; a < 8; ++a)
#pragma unroll
        for (int m = 0; m < 2; ++m)
            accO[a][m] = v8f{0.f, 0.f, 0.f, 0.f, 0.f, 0.f, 0.f, 0.f};

    for (int i0 = 0; i0 < kI; i0 += IC) {
        // ================= stage 1: a1 / a3 over K = H =================
        v8f acc1[2], acc3[2];
#pragma unroll
        for (int m = 0; m < 2; ++m) {
            acc1[m] = v8f{0.f, 0.f, 0.f, 0.f, 0.f, 0.f, 0.f, 0.f};
            acc3[m] = v8f{0.f, 0.f, 0.f, 0.f, 0.f, 0.f, 0.f, 0.f};
        }

        for (int kk = 0; kk < kH; kk += KC) {
            __syncthreads();                   // scratch region reuse fence

            // stage x tile [32][64] (bf16, row-major)
            {
                const int r  = tid >> 3;
                const int cs = (tid & 7) * 8;
                const float* xp = x + (size_t)tokS[r] * kH + kk + cs;
                const float4 a = *(const float4*)xp;
                const float4 b = *(const float4*)(xp + 4);
                float f[8] = {a.x, a.y, a.z, a.w, b.x, b.y, b.z, b.w};
                *(uint4*)(SM + XS_OFF + r * XS_STRIDE + 2 * cs) = pack8(f);
            }
            // stage w1/w3 K-transposed [c in 0..127][k in 0..63] (bf16)
            {
                const int c = tid & 127;
                const float* wp = (tid < 128) ? w1e : w3e;
                unsigned char* dst = SM + ((tid < 128) ? W1_OFF : W3_OFF);
                if (kk + KC < kH)
                    __builtin_prefetch(wp + (size_t)(kk + KC) * kI + i0 + c, 0, 1);
#pragma unroll
                for (int j = 0; j < 8; ++j) {
                    float f[8];
#pragma unroll
                    for (int kj = 0; kj < 8; ++kj)
                        f[kj] = wp[(size_t)(kk + j * 8 + kj) * kI + i0 + c];
                    *(uint4*)(dst + c * WT_STRIDE + 2 * (j * 8)) = pack8(f);
                }
            }
            __syncthreads();

            // each wave owns 16 I-cols of the chunk: c0 = wave*16
#pragma unroll
            for (int ks = 0; ks < 2; ++ks) {
                const v16bf b1 = load_bt(SM + W1_OFF, WT_STRIDE, wave * 16, ks * 32, lane);
                const v16bf b3 = load_bt(SM + W3_OFF, WT_STRIDE, wave * 16, ks * 32, lane);
#pragma unroll
                for (int mt = 0; mt < 2; ++mt) {
                    const v16bf a = load_a(SM + XS_OFF, XS_STRIDE, mt * 16, ks * 32, lane);
                    acc1[mt] = wmma_bf16(a, b1, acc1[mt]);
                    acc3[mt] = wmma_bf16(a, b3, acc3[mt]);
                }
            }
        }

        // h = silu(a1) * a3 -> hS (bf16)  [C/D layout: M = v + 8*half, N = ln]
#pragma unroll
        for (int mt = 0; mt < 2; ++mt)
#pragma unroll
            for (int v = 0; v < 8; ++v) {
                const float a1 = acc1[mt][v];
                const float a3 = acc3[mt][v];
                const float hv = (a1 / (1.f + __expf(-a1))) * a3;
                const int row = mt * 16 + lh * 8 + v;
                const int col = wave * 16 + ln;
                *(unsigned short*)(SM + HS_OFF + row * HS_STRIDE + 2 * col) =
                    bf1(hv);
            }

        // ================= stage 2: out += h @ w2 =================
#pragma unroll
        for (int half = 0; half < 2; ++half) {
#pragma unroll
            for (int ks2 = 0; ks2 < IC; ks2 += 32) {
                __syncthreads();               // also fences hS writes / w2 reuse
                // stage w2 K-transposed: [c in 0..511][k in 0..31] (bf16)
#pragma unroll
                for (int j = 0; j < 8; ++j) {
                    const int c  = (tid & 255) + (j & 1) * 256;
                    const int kg = j >> 1;     // 0..3 (8 k's each)
                    float f[8];
#pragma unroll
                    for (int kj = 0; kj < 8; ++kj)
                        f[kj] = w2e[(size_t)(i0 + ks2 + kg * 8 + kj) * kH +
                                    half * 512 + c];
                    *(uint4*)(SM + W2_OFF + c * W2_STRIDE + 2 * (kg * 8)) = pack8(f);
                }
                __syncthreads();

                // wave owns 64 cols of this half: 4 N-tiles
#pragma unroll
                for (int nt = 0; nt < 4; ++nt) {
                    const v16bf b2 =
                        load_bt(SM + W2_OFF, W2_STRIDE, wave * 64 + nt * 16, 0, lane);
#pragma unroll
                    for (int mt = 0; mt < 2; ++mt) {
                        const v16bf a =
                            load_a(SM + HS_OFF, HS_STRIDE, mt * 16, ks2, lane);
                        accO[half * 4 + nt][mt] =
                            wmma_bf16(a, b2, accO[half * 4 + nt][mt]);
                    }
                }
            }
        }
    }

    // ================= epilogue: out[tok] += weight * acc =================
#pragma unroll
    for (int an = 0; an < 8; ++an)
#pragma unroll
        for (int mt = 0; mt < 2; ++mt)
#pragma unroll
            for (int v = 0; v < 8; ++v) {
                const int row = mt * 16 + lh * 8 + v;
                const float wgt = twtS[row];
                if (wgt != 0.f) {
                    const int tok = tokS[row];
                    const int col = (an >> 2) * 512 + wave * 64 + (an & 3) * 16 + ln;
                    atomicAdd(out + (size_t)tok * kH + col, wgt * accO[an][mt][v]);
                }
            }
}

// ---------------------------------------------------------------------------
extern "C" void kernel_launch(void* const* d_in, const int* in_sizes, int n_in,
                              void* d_out, int out_size, void* d_ws, size_t ws_size,
                              hipStream_t stream) {
    const float* x  = (const float*)d_in[0];   // [4,4096,1024]
    const float* wr = (const float*)d_in[1];   // [1024,8]
    const float* w1 = (const float*)d_in[2];   // [8,1024,2816]
    const float* w3 = (const float*)d_in[3];   // [8,1024,2816]
    const float* w2 = (const float*)d_in[4];   // [8,2816,1024]
    float* out = (float*)d_out;

    unsigned char* ws = (unsigned char*)d_ws;
    int*   cnt   = (int*)ws;                                  // 8 ints
    int*   lists = (int*)(ws + 256);                          // [E][T]
    float* lw    = (float*)(ws + 256 + (size_t)kE * kT * 4);  // [E][T]

    moe_zero_kernel<<<2048, 256, 0, stream>>>(out, cnt);
    moe_router_kernel<<<kT / 8, 256, 0, stream>>>(x, wr, cnt, lists, lw);
    moe_expert_kernel<<<kE * TILES_PER_E, 256, 0, stream>>>(
        x, w1, w3, w2, cnt, lists, lw, out);
}
